// LigerLMHeadSimPO_68307159875594
// MI455X (gfx1250) — compile-verified, matrix operand order checked by reading, not run on hardware
//
#include <hip/hip_runtime.h>
#include <cstdint>
#include <cstddef>

#define HDIM 1024
#define VDIM 128000
#define BT   2048          // B*T tokens
#define TT   512           // tokens per sequence
#define NB   4             // batch
#define NCHUNK 500         // vocab chunks
#define CCOLS  256         // cols per chunk (16 n-tiles)
#define MBLK   256         // token rows per workgroup (16 row-tiles)
#define LBUF   4096        // uints per LDS tile buffer (256 rows x 16 uints)
#define LBUFB  (LBUF * 4)  // bytes per LDS tile buffer (16 KB)
#define IGNORE_INDEX (-100)
#define BETA  0.1f
#define GAMMA 0.5f

typedef __attribute__((ext_vector_type(16))) __bf16 v16bf;
typedef __attribute__((ext_vector_type(8)))  float  v8f;
typedef __attribute__((ext_vector_type(4)))  unsigned tdm_u32x4;
typedef __attribute__((ext_vector_type(8)))  int      tdm_i32x8;
typedef __attribute__((ext_vector_type(4)))  int      tdm_i32x4;

#if __has_builtin(__builtin_amdgcn_tensor_load_to_lds)
#define HAVE_TDM 1
#else
#define HAVE_TDM 0
#endif

struct U4x2 { uint4 lo, hi; };

__device__ __forceinline__ unsigned pack2_bf16(float a, float b) {
    unsigned ua = __float_as_uint(a);
    unsigned ub = __float_as_uint(b);
    ua = (ua + 0x7FFFu + ((ua >> 16) & 1u)) >> 16;   // round-to-nearest-even
    ub = (ub + 0x7FFFu + ((ub >> 16) & 1u)) >> 16;
    return ua | (ub << 16);
}

__device__ __forceinline__ void cvt_store16(const float4* f, unsigned* dst) {
    uint4 p0 = make_uint4(pack2_bf16(f[0].x, f[0].y), pack2_bf16(f[0].z, f[0].w),
                          pack2_bf16(f[1].x, f[1].y), pack2_bf16(f[1].z, f[1].w));
    uint4 p1 = make_uint4(pack2_bf16(f[2].x, f[2].y), pack2_bf16(f[2].z, f[2].w),
                          pack2_bf16(f[3].x, f[3].y), pack2_bf16(f[3].z, f[3].w));
    ((uint4*)dst)[0] = p0;
    ((uint4*)dst)[1] = p1;
}

#if HAVE_TDM
// TDM D# for one 2-D tile: tile_dim0=32 bf16 along K (contiguous),
// tile_dim1=256 rows, row stride = HDIM elements. Groups 2/3 unused (2-D).
// This toolchain exposes the 6-arg builtin:
//   (u32x4 g0, i32x8 g1, i32x4 g2, i32x4 g3, i32x8, i32 cpol)
__device__ __forceinline__ void tdm_load_tile(unsigned lds_off,
                                              unsigned long long gaddr,
                                              int tensor_dim1) {
    tdm_u32x4 g0;
    g0.x = 1u;                                      // count=1, user descriptor
    g0.y = lds_off;                                 // lds_addr (bytes)
    g0.z = (unsigned)(gaddr & 0xFFFFFFFFull);       // global_addr[31:0] -> bits 95:64
    g0.w = (unsigned)((gaddr >> 32) & 0x01FFFFFFull) | (2u << 30); // addr[56:32]|type=2
    tdm_i32x8 g1;
    g1[0] = 0x00010000;                             // data_size=1 (2 bytes), no mcast
    g1[1] = (int)((HDIM & 0xFFFF) << 16);           // tensor_dim0[15:0] @ bits 63:48
    g1[2] = (int)(((unsigned)HDIM >> 16) & 0xFFFF)  // tensor_dim0[31:16]
          | (int)((tensor_dim1 & 0xFFFF) << 16);    // tensor_dim1[15:0]
    g1[3] = (int)(((unsigned)tensor_dim1 >> 16) & 0xFFFF)
          | (32 << 16);                             // tile_dim0 = 32
    g1[4] = 256;                                    // tile_dim1=256, tile_dim2=0
    g1[5] = HDIM;                                   // tensor_dim0_stride lo32
    g1[6] = 0;
    g1[7] = 0;
    tdm_i32x4 zz4 = {0, 0, 0, 0};
    tdm_i32x8 zz8 = {0, 0, 0, 0, 0, 0, 0, 0};
    __builtin_amdgcn_tensor_load_to_lds(g0, g1, zz4, zz4, zz8, 0);
}
#endif

// ---------------------------------------------------------------------------
// fp32 -> bf16 bulk converter (8 elements / thread)
// ---------------------------------------------------------------------------
__global__ __launch_bounds__(256)
void cvt_f32_bf16(const float* __restrict__ src, unsigned* __restrict__ dst,
                  long long n8) {
    long long i = (long long)blockIdx.x * 256 + threadIdx.x;
    if (i >= n8) return;
    const float4* s = (const float4*)(src + i * 8);
    float4 a = s[0], b = s[1];
    uint4 p = make_uint4(pack2_bf16(a.x, a.y), pack2_bf16(a.z, a.w),
                         pack2_bf16(b.x, b.y), pack2_bf16(b.z, b.w));
    ((uint4*)dst)[i] = p;
}

// ---------------------------------------------------------------------------
// GEMM + partial LSE, bf16 inputs. LDS tiles double-buffered; staged by the
// Tensor Data Mover (one wave issues 2 tensor_load_to_lds per k-step, waits
// on TENSORcnt, workgroup barrier hands the tile to all 16 waves).
// ---------------------------------------------------------------------------
__global__ __launch_bounds__(512)
void simpo_gemm_lse_partial_bf16(const unsigned short* __restrict__ x2,
                                 const unsigned short* __restrict__ w2,
                                 float* __restrict__ pmax,
                                 float* __restrict__ psum) {
    __shared__ __align__(16) unsigned Xls[2 * LBUF];
    __shared__ __align__(16) unsigned Wls[2 * LBUF];
    __shared__ float redm[MBLK][4];
    __shared__ float reds[MBLK][4];

    const int chunk   = blockIdx.x;
    const int rowbase = blockIdx.y * MBLK;
    const int colbase = chunk * CCOLS;
    const int tid  = threadIdx.x;
    const int lane = tid & 31;
    const int wave = tid >> 5;
    const int rtg  = wave & 3;
    const int ntg  = wave >> 2;
    const int half = lane >> 4;
    const int l15  = lane & 15;

    v8f acc[4][4];
    #pragma unroll
    for (int ai = 0; ai < 4; ++ai)
        #pragma unroll
        for (int bi = 0; bi < 4; ++bi) acc[ai][bi] = {};

    const unsigned* abase = &Xls[(rtg * 64 + l15) * 16 + half * 4];
    const unsigned* bbase = &Wls[(ntg * 64 + l15) * 16 + half * 4];

#if HAVE_TDM
    const unsigned xls0 = (unsigned)(unsigned long long)&Xls[0];
    const unsigned wls0 = (unsigned)(unsigned long long)&Wls[0];
    const unsigned long long xg =
        (unsigned long long)(const void*)x2 + (unsigned long long)rowbase * HDIM * 2;
    const unsigned long long wg =
        (unsigned long long)(const void*)w2 + (unsigned long long)colbase * HDIM * 2;
    if (wave == 0) {                       // prologue: slice k=0 -> buffer 0
        tdm_load_tile(xls0, xg, BT);
        tdm_load_tile(wls0, wg, VDIM);
        __builtin_amdgcn_s_wait_tensorcnt(0);
    }
#else
    const int ldr = tid >> 1;
    const int ldh = (tid & 1) * 16;
    const unsigned short* xsrc = x2 + (size_t)(rowbase + ldr) * HDIM + ldh;
    const unsigned short* wsrc = w2 + (size_t)(colbase + ldr) * HDIM + ldh;
    unsigned* xdst = &Xls[ldr * 16 + (ldh >> 1)];
    unsigned* wdst = &Wls[ldr * 16 + (ldh >> 1)];
    {
        uint4 vx0 = ((const uint4*)xsrc)[0], vx1 = ((const uint4*)xsrc)[1];
        uint4 vw0 = ((const uint4*)wsrc)[0], vw1 = ((const uint4*)wsrc)[1];
        ((uint4*)xdst)[0] = vx0; ((uint4*)xdst)[1] = vx1;
        ((uint4*)wdst)[0] = vw0; ((uint4*)wdst)[1] = vw1;
    }
#endif
    __syncthreads();

    for (int k0 = 0; k0 < HDIM; k0 += 32) {
        const int cur = (k0 >> 5) & 1;
        const int nxt = cur ^ 1;
        const bool have_next = (k0 + 32 < HDIM);

#if HAVE_TDM
        if (wave == 0 && have_next) {      // DMA next slice into other buffer
            tdm_load_tile(xls0 + nxt * LBUFB, xg + (unsigned long long)(k0 + 32) * 2, BT);
            tdm_load_tile(wls0 + nxt * LBUFB, wg + (unsigned long long)(k0 + 32) * 2, VDIM);
        }
#else
        uint4 vx0, vx1, vw0, vw1;
        if (have_next) {
            const uint4* sx = (const uint4*)(xsrc + k0 + 32);
            const uint4* sw = (const uint4*)(wsrc + k0 + 32);
            vx0 = sx[0]; vx1 = sx[1]; vw0 = sw[0]; vw1 = sw[1];
        }
#endif

        // 16 WMMA on the CURRENT buffer
        U4x2 a_raw[4];
        #pragma unroll
        for (int ai = 0; ai < 4; ++ai) {
            const unsigned* ap = abase + cur * LBUF + ai * 256;
            a_raw[ai].lo = *(const uint4*)ap;
            a_raw[ai].hi = *(const uint4*)(ap + 8);
        }
        U4x2 b_cur;
        {
            const unsigned* bp = bbase + cur * LBUF;
            b_cur.lo = *(const uint4*)bp;
            b_cur.hi = *(const uint4*)(bp + 8);
        }
        #pragma unroll
        for (int bi = 0; bi < 4; ++bi) {
            U4x2 b_nxt = b_cur;
            if (bi < 3) {
                const unsigned* bp = bbase + cur * LBUF + (bi + 1) * 256;
                b_nxt.lo = *(const uint4*)bp;
                b_nxt.hi = *(const uint4*)(bp + 8);
            }
            v16bf bv = __builtin_bit_cast(v16bf, b_cur);
            #pragma unroll
            for (int ai = 0; ai < 4; ++ai) {
                v16bf av = __builtin_bit_cast(v16bf, a_raw[ai]);
                acc[ai][bi] = __builtin_amdgcn_wmma_f32_16x16x32_bf16(
                    false, av, false, bv, (short)0, acc[ai][bi], false, false);
            }
            b_cur = b_nxt;
        }

#if HAVE_TDM
        if (wave == 0 && have_next)
            __builtin_amdgcn_s_wait_tensorcnt(0);   // next buffer filled
#else
        if (have_next) {
            ((uint4*)(xdst + nxt * LBUF))[0] = vx0;
            ((uint4*)(xdst + nxt * LBUF))[1] = vx1;
            ((uint4*)(wdst + nxt * LBUF))[0] = vw0;
            ((uint4*)(wdst + nxt * LBUF))[1] = vw1;
        }
#endif
        __syncthreads();
    }

    // Row-wise (max, sum-exp) over this wave's 64-col group.
    #pragma unroll
    for (int ai = 0; ai < 4; ++ai) {
        #pragma unroll
        for (int i = 0; i < 8; ++i) {
            float mi = acc[ai][0][i];
            #pragma unroll
            for (int bi = 1; bi < 4; ++bi) mi = fmaxf(mi, acc[ai][bi][i]);
            #pragma unroll
            for (int off = 1; off < 16; off <<= 1)
                mi = fmaxf(mi, __shfl_xor(mi, off, 32));
            float si = 0.0f;
            #pragma unroll
            for (int bi = 0; bi < 4; ++bi) si += __expf(acc[ai][bi][i] - mi);
            #pragma unroll
            for (int off = 1; off < 16; off <<= 1)
                si += __shfl_xor(si, off, 32);
            if (l15 == 0) {
                int rl = rtg * 64 + ai * 16 + half * 8 + i;
                redm[rl][ntg] = mi;
                reds[rl][ntg] = si;
            }
        }
    }
    __syncthreads();

    if (tid < MBLK) {
        float M = fmaxf(fmaxf(redm[tid][0], redm[tid][1]),
                        fmaxf(redm[tid][2], redm[tid][3]));
        float S = 0.0f;
        #pragma unroll
        for (int j = 0; j < 4; ++j) S += reds[tid][j] * __expf(redm[tid][j] - M);
        pmax[(size_t)chunk * BT + rowbase + tid] = M;
        psum[(size_t)chunk * BT + rowbase + tid] = S;
    }
}

// ---------------------------------------------------------------------------
// Fallback GEMM (fp32 inputs, in-kernel conversion) for small workspaces —
// identical to the round-3 pipelined kernel.
// ---------------------------------------------------------------------------
__global__ __launch_bounds__(512)
void simpo_gemm_lse_partial_f32(const float* __restrict__ x,
                                const float* __restrict__ W,
                                float* __restrict__ pmax,
                                float* __restrict__ psum) {
    __shared__ __align__(16) unsigned Xls[2 * LBUF];
    __shared__ __align__(16) unsigned Wls[2 * LBUF];
    __shared__ float redm[MBLK][4];
    __shared__ float reds[MBLK][4];

    const int chunk   = blockIdx.x;
    const int rowbase = blockIdx.y * MBLK;
    const int colbase = chunk * CCOLS;
    const int tid  = threadIdx.x;
    const int lane = tid & 31;
    const int wave = tid >> 5;
    const int rtg  = wave & 3;
    const int ntg  = wave >> 2;
    const int half = lane >> 4;
    const int l15  = lane & 15;

    v8f acc[4][4];
    #pragma unroll
    for (int ai = 0; ai < 4; ++ai)
        #pragma unroll
        for (int bi = 0; bi < 4; ++bi) acc[ai][bi] = {};

    const int ldr = tid >> 1;
    const int ldh = (tid & 1) * 16;
    const float* xsrc = x + (size_t)(rowbase + ldr) * HDIM + ldh;
    const float* wsrc = W + (size_t)(colbase + ldr) * HDIM + ldh;
    unsigned* xdst = &Xls[ldr * 16 + (ldh >> 1)];
    unsigned* wdst = &Wls[ldr * 16 + (ldh >> 1)];

    const unsigned* abase = &Xls[(rtg * 64 + l15) * 16 + half * 4];
    const unsigned* bbase = &Wls[(ntg * 64 + l15) * 16 + half * 4];

    {
        float4 fx[4], fw[4];
        const float4* sx = (const float4*)xsrc;
        const float4* sw = (const float4*)wsrc;
        fx[0] = sx[0]; fx[1] = sx[1]; fx[2] = sx[2]; fx[3] = sx[3];
        fw[0] = sw[0]; fw[1] = sw[1]; fw[2] = sw[2]; fw[3] = sw[3];
        cvt_store16(fx, xdst);
        cvt_store16(fw, wdst);
    }
    __syncthreads();

    for (int k0 = 0; k0 < HDIM; k0 += 32) {
        const int cur = (k0 >> 5) & 1;
        const int nxt = cur ^ 1;
        const bool have_next = (k0 + 32 < HDIM);

        float4 fx[4], fw[4];
        if (have_next) {
            const float4* sx = (const float4*)(xsrc + k0 + 32);
            const float4* sw = (const float4*)(wsrc + k0 + 32);
            fx[0] = sx[0]; fx[1] = sx[1]; fx[2] = sx[2]; fx[3] = sx[3];
            fw[0] = sw[0]; fw[1] = sw[1]; fw[2] = sw[2]; fw[3] = sw[3];
            if (k0 + 64 < HDIM) {
                __builtin_prefetch(xsrc + k0 + 64, 0, 1);
                __builtin_prefetch(wsrc + k0 + 64, 0, 1);
            }
        }

        U4x2 a_raw[4];
        #pragma unroll
        for (int ai = 0; ai < 4; ++ai) {
            const unsigned* ap = abase + cur * LBUF + ai * 256;
            a_raw[ai].lo = *(const uint4*)ap;
            a_raw[ai].hi = *(const uint4*)(ap + 8);
        }
        U4x2 b_cur;
        {
            const unsigned* bp = bbase + cur * LBUF;
            b_cur.lo = *(const uint4*)bp;
            b_cur.hi = *(const uint4*)(bp + 8);
        }
        #pragma unroll
        for (int bi = 0; bi < 4; ++bi) {
            U4x2 b_nxt = b_cur;
            if (bi < 3) {
                const unsigned* bp = bbase + cur * LBUF + (bi + 1) * 256;
                b_nxt.lo = *(const uint4*)bp;
                b_nxt.hi = *(const uint4*)(bp + 8);
            }
            v16bf bv = __builtin_bit_cast(v16bf, b_cur);
            #pragma unroll
            for (int ai = 0; ai < 4; ++ai) {
                v16bf av = __builtin_bit_cast(v16bf, a_raw[ai]);
                acc[ai][bi] = __builtin_amdgcn_wmma_f32_16x16x32_bf16(
                    false, av, false, bv, (short)0, acc[ai][bi], false, false);
            }
            b_cur = b_nxt;
        }

        if (have_next) {
            cvt_store16(fx, xdst + nxt * LBUF);
            cvt_store16(fw, wdst + nxt * LBUF);
        }
        __syncthreads();
    }

    #pragma unroll
    for (int ai = 0; ai < 4; ++ai) {
        #pragma unroll
        for (int i = 0; i < 8; ++i) {
            float mi = acc[ai][0][i];
            #pragma unroll
            for (int bi = 1; bi < 4; ++bi) mi = fmaxf(mi, acc[ai][bi][i]);
            #pragma unroll
            for (int off = 1; off < 16; off <<= 1)
                mi = fmaxf(mi, __shfl_xor(mi, off, 32));
            float si = 0.0f;
            #pragma unroll
            for (int bi = 0; bi < 4; ++bi) si += __expf(acc[ai][bi][i] - mi);
            #pragma unroll
            for (int off = 1; off < 16; off <<= 1)
                si += __shfl_xor(si, off, 32);
            if (l15 == 0) {
                int rl = rtg * 64 + ai * 16 + half * 8 + i;
                redm[rl][ntg] = mi;
                reds[rl][ntg] = si;
            }
        }
    }
    __syncthreads();

    if (tid < MBLK) {
        float M = fmaxf(fmaxf(redm[tid][0], redm[tid][1]),
                        fmaxf(redm[tid][2], redm[tid][3]));
        float S = 0.0f;
        #pragma unroll
        for (int j = 0; j < 4; ++j) S += reds[tid][j] * __expf(redm[tid][j] - M);
        pmax[(size_t)chunk * BT + rowbase + tid] = M;
        psum[(size_t)chunk * BT + rowbase + tid] = S;
    }
}

// ---------------------------------------------------------------------------
// Kernel: one wave per token — merge chunk partials into LSE, fp32 dot with
// the label row of W, masked per-token logp.
// ---------------------------------------------------------------------------
__global__ __launch_bounds__(256)
void simpo_logp(const float* __restrict__ x, const float* __restrict__ W,
                const int* __restrict__ y,
                const float* __restrict__ pmax, const float* __restrict__ psum,
                float* __restrict__ logp) {
    const int lane  = threadIdx.x & 31;
    const int wave  = threadIdx.x >> 5;
    const int token = blockIdx.x * 8 + wave;
    if (token >= BT) return;

    int label = y[token];
    int valid = (label != IGNORE_INDEX);
    int cl = label < 0 ? 0 : (label >= VDIM ? VDIM - 1 : label);

    const float* xr = x + (size_t)token * HDIM;
    const float* wr = W + (size_t)cl * HDIM;
    float dot = 0.0f;
    for (int k = lane; k < HDIM; k += 32) dot = fmaf(xr[k], wr[k], dot);
    #pragma unroll
    for (int off = 16; off; off >>= 1) dot += __shfl_xor(dot, off, 32);

    float M = -3.4e38f;
    for (int c = lane; c < NCHUNK; c += 32)
        M = fmaxf(M, pmax[(size_t)c * BT + token]);
    #pragma unroll
    for (int off = 16; off; off >>= 1) M = fmaxf(M, __shfl_xor(M, off, 32));

    float Z = 0.0f;
    for (int c = lane; c < NCHUNK; c += 32)
        Z += psum[(size_t)c * BT + token] * __expf(pmax[(size_t)c * BT + token] - M);
    #pragma unroll
    for (int off = 16; off; off >>= 1) Z += __shfl_xor(Z, off, 32);

    if (lane == 0) {
        float lse = M + __logf(Z);
        logp[token] = valid ? (dot - lse) : 0.0f;
    }
}

// ---------------------------------------------------------------------------
// Kernel: masked per-sequence means + SimPO pair loss -> scalar.
// ---------------------------------------------------------------------------
__global__ __launch_bounds__(256)
void simpo_final(const float* __restrict__ logp, const int* __restrict__ y,
                 float* __restrict__ out) {
    __shared__ float ssum[NB];
    __shared__ int   scnt[NB];
    const int tid = threadIdx.x;
    if (tid < NB) { ssum[tid] = 0.0f; scnt[tid] = 0; }
    __syncthreads();
    for (int i = tid; i < BT; i += 256) {
        int b = i / TT;
        atomicAdd(&ssum[b], logp[i]);                 // ds_add_f32
        if (y[i] != IGNORE_INDEX) atomicAdd(&scnt[b], 1);
    }
    __syncthreads();
    if (tid == 0) {
        float avg[NB];
        #pragma unroll
        for (int b = 0; b < NB; ++b)
            avg[b] = ssum[b] / fmaxf((float)scnt[b], 1.0f);
        float loss = 0.0f;
        #pragma unroll
        for (int p = 0; p < NB / 2; ++p) {
            float d = BETA * (avg[p] - avg[p + NB / 2]) - GAMMA;
            float z = -d;   // loss = softplus(-d) (label smoothing 0)
            loss += (z > 0.0f) ? z + log1pf(expf(-z)) : log1pf(expf(z));
        }
        out[0] = loss / (float)(NB / 2);
    }
}

extern "C" void kernel_launch(void* const* d_in, const int* in_sizes, int n_in,
                              void* d_out, int out_size, void* d_ws, size_t ws_size,
                              hipStream_t stream) {
    const float* x = (const float*)d_in[0];
    const int*   y = (const int*)d_in[1];
    const float* W = (const float*)d_in[2];
    float* out = (float*)d_out;

    const size_t partials = (size_t)NCHUNK * BT;
    float* pmax = (float*)d_ws;
    float* psum = pmax + partials;
    float* logp = psum + partials;
    unsigned short* x2 = (unsigned short*)(logp + BT);
    unsigned short* w2 = x2 + (size_t)BT * HDIM;

    const size_t need = (2 * partials + BT) * sizeof(float)
                      + ((size_t)BT * HDIM + (size_t)VDIM * HDIM) * 2;

    dim3 g1(NCHUNK, BT / MBLK);                       // 500 x 8 workgroups
    if (ws_size >= need) {
        const long long xn8 = (long long)BT * HDIM / 8;
        const long long wn8 = (long long)VDIM * HDIM / 8;
        cvt_f32_bf16<<<(unsigned)((xn8 + 255) / 256), 256, 0, stream>>>(
            x, (unsigned*)x2, xn8);
        cvt_f32_bf16<<<(unsigned)((wn8 + 255) / 256), 256, 0, stream>>>(
            W, (unsigned*)w2, wn8);
        simpo_gemm_lse_partial_bf16<<<g1, 512, 0, stream>>>(x2, w2, pmax, psum);
    } else {
        simpo_gemm_lse_partial_f32<<<g1, 512, 0, stream>>>(x, W, pmax, psum);
    }
    simpo_logp<<<BT / 8, 256, 0, stream>>>(x, W, y, pmax, psum, logp);
    simpo_final<<<1, 256, 0, stream>>>(logp, y, out);
}
